// QIGAT_39608188404031
// MI455X (gfx1250) — matrix operand order repaired and checked by reading, not compiled, not run on hardware
//
#include <hip/hip_runtime.h>
#include <hip/hip_bf16.h>

#define IN_CH 165
#define HID 128
#define KPAD_C 384
#define QDIM 358
#define NPAIR 28
#define NEG_SLOPE 0.2f
#define MAX_K 384

typedef __attribute__((ext_vector_type(16))) __bf16 v16bf;
typedef __attribute__((ext_vector_type(8)))  __bf16 v8bf;
typedef __attribute__((ext_vector_type(8)))  float  v8f;

__device__ __constant__ int c_IU0[NPAIR] = {0,0,0,0,0,0,0,1,1,1,1,1,1,2,2,2,2,2,3,3,3,3,4,4,4,5,5,6};
__device__ __constant__ int c_IU1[NPAIR] = {1,2,3,4,5,6,7,2,3,4,5,6,7,3,4,5,6,7,4,5,6,7,5,6,7,6,7,7};

__device__ __forceinline__ float eluf(float x) { return x > 0.f ? x : (expf(x) - 1.f); }

__device__ __forceinline__ void atomicMaxF(float* a, float v) {
    if (v >= 0.f) atomicMax((int*)a, __float_as_int(v));
    else          atomicMin((unsigned int*)a, __float_as_uint(v));
}

// ---------------------------------------------------------------------------
// Repack f32 weights [Ksrc, M] (row-major) into bf16 WMMA B-fragment order:
// out[((tileN*(Kpad/32)+ks)*32 + lane)*16 + j] = W[ks*32 + (lane>=16)*16 + j][tileN*16 + (lane&15)]
// so each lane's 16-half B fragment is one contiguous 32B load in the GEMM.
// ---------------------------------------------------------------------------
__global__ void repack_kernel(const float* __restrict__ W, __bf16* __restrict__ out,
                              int Ksrc, int Kpad, int M) {
    int tid = blockIdx.x * blockDim.x + threadIdx.x;
    int total = (M >> 4) * (Kpad >> 5) * 32;
    if (tid >= total) return;
    int lane   = tid & 31;
    int rest   = tid >> 5;
    int ksteps = Kpad >> 5;
    int ks     = rest % ksteps;
    int tn     = rest / ksteps;
    int col    = (tn << 4) + (lane & 15);
    int kb     = (ks << 5) + ((lane >> 4) << 4);
    __bf16* o  = out + (size_t)tid * 16;
#pragma unroll
    for (int j = 0; j < 16; j++) {
        int kr  = kb + j;
        float v = (kr < Ksrc) ? W[(size_t)kr * M + col] : 0.f;
        o[j] = (__bf16)v;
    }
}

// ---------------------------------------------------------------------------
// Generic WMMA GEMM: 16 rows x M cols per block, one 16x16 tile per wave.
// QMAP=true: A is computed on the fly from x [N,165] (quantum feature map).
// flags: 1=bias  2=layernorm  4=elu.  outB (optional) mirrors outF in bf16.
// ---------------------------------------------------------------------------
template <bool QMAP>
__global__ __launch_bounds__(256) void wmma_gemm_kernel(
    const float* __restrict__ xin, const __bf16* __restrict__ Ain,
    const __bf16* __restrict__ Brep, const float* __restrict__ bias,
    const float* __restrict__ lng, const float* __restrict__ lnb,
    float* __restrict__ outF, __bf16* __restrict__ outB,
    int N, int K, int M, int flags) {
    __shared__ __bf16 sA[16 * MAX_K];
    __shared__ float  sC[16 * 128];
    __shared__ float  sRed[2][256];
    __shared__ float  sStats[2][16];

    const int tid = threadIdx.x;
    const int rowBase = blockIdx.x * 16;

    // ---- stage A tile (16 x K) into LDS ----
    int chunks = (16 * K) >> 3;
    for (int c = tid; c < chunks; c += blockDim.x) {
        int r = (c << 3) / K;
        int k0 = (c << 3) % K;
        int grow = rowBase + r;
        __bf16 tmp[8];
        if (QMAP) {
#pragma unroll
            for (int j = 0; j < 8; j++) {
                int k = k0 + j;
                float v = 0.f;
                if (grow < N) {
                    const float* xr = xin + (size_t)grow * IN_CH;
                    if (k < IN_CH)            v = cosf(xr[k]);
                    else if (k < 2 * IN_CH)   v = sinf(xr[k - IN_CH]);
                    else if (k < QDIM) {
                        int pIdx = k - 2 * IN_CH;
                        v = sinf(xr[c_IU0[pIdx]]) * sinf(xr[c_IU1[pIdx]]);
                    }
                }
                tmp[j] = (__bf16)v;
            }
        } else {
            if (grow < N) {
                const __bf16* src = Ain + (size_t)grow * K + k0;
#pragma unroll
                for (int j = 0; j < 8; j++) tmp[j] = src[j];
            } else {
#pragma unroll
                for (int j = 0; j < 8; j++) tmp[j] = (__bf16)0.f;
            }
        }
#pragma unroll
        for (int j = 0; j < 8; j++) sA[r * K + k0 + j] = tmp[j];
    }
    __syncthreads();

    const int wave = tid >> 5, lane = tid & 31;
    const int l15 = lane & 15, hi = lane >> 4;

    v8f acc = {0.f, 0.f, 0.f, 0.f, 0.f, 0.f, 0.f, 0.f};
    const int ksteps = K >> 5;
    const __bf16* bptr = Brep + ((size_t)(wave * ksteps) * 32 + lane) * 16;
    for (int ks = 0; ks < ksteps; ks++) {
        // A fragment: lane (l15,hi) holds K run [ks*32+hi*16, +16)
        const __bf16* ap = &sA[l15 * K + ks * 32 + hi * 16];
        v8bf alo = *(const v8bf*)(ap);
        v8bf ahi2 = *(const v8bf*)(ap + 8);
        v16bf afrag = __builtin_shufflevector(alo, ahi2, 0, 1, 2, 3, 4, 5, 6, 7,
                                              8, 9, 10, 11, 12, 13, 14, 15);
        v16bf bfrag = *(const v16bf*)(bptr + (size_t)ks * 512);
        acc = __builtin_amdgcn_wmma_f32_16x16x32_bf16(false, afrag, false, bfrag,
                                                      (short)0, acc, false, false);
    }

    // ---- spill C tile to LDS (lane l15 = col, vgpr i = row (+8 for hi lanes)) ----
    {
        int col = wave * 16 + l15;
        int rbase = hi * 8;
#pragma unroll
        for (int i = 0; i < 8; i++) sC[(rbase + i) * M + col] = acc[i];
    }
    __syncthreads();

    // ---- fused epilogue: bias / layernorm / elu / dual-precision store ----
    const bool hasBias = flags & 1, hasLN = flags & 2, hasELU = flags & 4;
    const int nc = M >> 4;
    const int rowsPerIter = blockDim.x >> 4;
    const int l = tid & 15;
    for (int r0 = tid >> 4; r0 < 16; r0 += rowsPerIter) {
        float vals[8];
        float s = 0.f;
        for (int k = 0; k < nc; k++) {
            float v = sC[r0 * M + l + (k << 4)];
            if (hasBias) v += bias[l + (k << 4)];
            vals[k] = v;
            s += v;
        }
        float mu = 0.f, rs = 1.f;
        if (hasLN) {  // only used with M=128 / 256 threads (single iteration)
            float ss = 0.f;
            for (int k = 0; k < nc; k++) ss += vals[k] * vals[k];
            sRed[0][r0 * 16 + l] = s;
            sRed[1][r0 * 16 + l] = ss;
            __syncthreads();
            if (l == 0) {
                float S = 0.f, SS = 0.f;
                for (int i = 0; i < 16; i++) { S += sRed[0][r0 * 16 + i]; SS += sRed[1][r0 * 16 + i]; }
                float m = S / (float)M;
                float var = SS / (float)M - m * m;
                sStats[0][r0] = m;
                sStats[1][r0] = rsqrtf(var + 1e-5f);
            }
            __syncthreads();
            mu = sStats[0][r0];
            rs = sStats[1][r0];
        }
        int grow = rowBase + r0;
        if (grow < N) {
            for (int k = 0; k < nc; k++) {
                int col = l + (k << 4);
                float v = vals[k];
                if (hasLN) v = (v - mu) * rs * lng[col] + lnb[col];
                if (hasELU) v = eluf(v);
                outF[(size_t)grow * M + col] = v;
                if (outB) outB[(size_t)grow * M + col] = (__bf16)v;
            }
        }
    }
}

// ---------------------------------------------------------------------------
// Attention scores: al_s/al_d[n,h] = <hp[n, h*16:+16], a_src/dst[h]>
// ---------------------------------------------------------------------------
__global__ void att_scores_kernel(const float* __restrict__ hp,
                                  const float* __restrict__ asrc,
                                  const float* __restrict__ adst,
                                  float* __restrict__ als, float* __restrict__ ald, int N) {
    int tid = blockIdx.x * blockDim.x + threadIdx.x;
    if (tid >= N * 8) return;
    int n = tid >> 3, h = tid & 7;
    const float* v = hp + (size_t)n * HID + (h << 4);
    float s = 0.f, d = 0.f;
#pragma unroll
    for (int c = 0; c < 16; c++) {
        float hv = v[c];
        s += hv * asrc[(h << 4) + c];
        d += hv * adst[(h << 4) + c];
    }
    als[tid] = s;
    ald[tid] = d;
}

__global__ void gat_init_kernel(float* agg, float* denom, float* emax, int N) {
    int tid = blockIdx.x * blockDim.x + threadIdx.x;
    int total = N * (HID + 16);
    if (tid >= total) return;
    if (tid < N * HID) agg[tid] = 0.f;
    else {
        int t = tid - N * HID;
        if (t < N * 8) denom[t] = 0.f;
        else           emax[t - N * 8] = -INFINITY;
    }
}

__global__ void edge_logits_max_kernel(const int* __restrict__ ei,
                                       const float* __restrict__ als,
                                       const float* __restrict__ ald,
                                       float* __restrict__ emax, float* __restrict__ ebuf, int E) {
    int tid = blockIdx.x * blockDim.x + threadIdx.x;
    if (tid >= E * 8) return;
    int e = tid >> 3, h = tid & 7;
    int s = ei[e], d = ei[E + e];
    float el = als[s * 8 + h] + ald[d * 8 + h];
    el = el > 0.f ? el : NEG_SLOPE * el;
    ebuf[tid] = el;
    atomicMaxF(&emax[d * 8 + h], el);
}

__global__ void edge_exp_sum_kernel(const int* __restrict__ ei,
                                    const float* __restrict__ emax,
                                    float* __restrict__ denom, float* __restrict__ ebuf, int E) {
    int tid = blockIdx.x * blockDim.x + threadIdx.x;
    if (tid >= E * 8) return;
    int e = tid >> 3, h = tid & 7;
    int d = ei[E + e];
    float ee = expf(ebuf[tid] - emax[d * 8 + h]);
    ebuf[tid] = ee;
    atomicAdd(&denom[d * 8 + h], ee);
}

__global__ void edge_aggregate_kernel(const int* __restrict__ ei,
                                      const float* __restrict__ hp,
                                      const float* __restrict__ ebuf,
                                      const float* __restrict__ denom,
                                      float* __restrict__ agg, int E) {
    int tid = blockIdx.x * blockDim.x + threadIdx.x;
    if (tid >= E * 8) return;
    int e = tid >> 3, h = tid & 7;
    int s = ei[e], d = ei[E + e];
    float alpha = ebuf[tid] / (denom[d * 8 + h] + 1e-16f);
    const float* hs = hp + (size_t)s * HID + (h << 4);
    float* ad = agg + (size_t)d * HID + (h << 4);
#pragma unroll
    for (int c = 0; c < 16; c++) atomicAdd(&ad[c], hs[c] * alpha);
}

// ---------------------------------------------------------------------------
// GAT finalize: h = elu(LN(agg + bias)) + h  (residual), refresh bf16 mirror.
// ---------------------------------------------------------------------------
__global__ __launch_bounds__(256) void gat_finalize_kernel(
    const float* __restrict__ agg, const float* __restrict__ bias,
    const float* __restrict__ lng, const float* __restrict__ lnb,
    float* __restrict__ h, __bf16* __restrict__ hbf, int N) {
    __shared__ float sRed[2][256];
    __shared__ float sStats[2][16];
    int tid = threadIdx.x, r0 = tid >> 4, l = tid & 15;
    int grow = blockIdx.x * 16 + r0;
    float vals[8];
    float s = 0.f, ss = 0.f;
#pragma unroll
    for (int k = 0; k < 8; k++) {
        int col = l + (k << 4);
        float v = (grow < N ? agg[(size_t)grow * HID + col] : 0.f) + bias[col];
        vals[k] = v;
        s += v;
        ss += v * v;
    }
    sRed[0][tid] = s;
    sRed[1][tid] = ss;
    __syncthreads();
    if (l == 0) {
        float S = 0.f, SS = 0.f;
        for (int i = 0; i < 16; i++) { S += sRed[0][r0 * 16 + i]; SS += sRed[1][r0 * 16 + i]; }
        float m = S / 128.f;
        float var = SS / 128.f - m * m;
        sStats[0][r0] = m;
        sStats[1][r0] = rsqrtf(var + 1e-5f);
    }
    __syncthreads();
    float mu = sStats[0][r0], rs = sStats[1][r0];
    if (grow < N) {
#pragma unroll
        for (int k = 0; k < 8; k++) {
            int col = l + (k << 4);
            float v = (vals[k] - mu) * rs * lng[col] + lnb[col];
            v = eluf(v) + h[(size_t)grow * HID + col];
            h[(size_t)grow * HID + col] = v;
            hbf[(size_t)grow * HID + col] = (__bf16)v;
        }
    }
}

__global__ void mlp2_kernel(const float* __restrict__ z, const float* __restrict__ W2,
                            const float* __restrict__ b2, float* __restrict__ out, int N) {
    int n = blockIdx.x * blockDim.x + threadIdx.x;
    if (n >= N) return;
    const float* zr = z + (size_t)n * 64;
    float a0 = b2[0], a1 = b2[1];
#pragma unroll 8
    for (int k = 0; k < 64; k++) {
        float v = zr[k];
        a0 += v * W2[2 * k];
        a1 += v * W2[2 * k + 1];
    }
    out[2 * n] = a0;
    out[2 * n + 1] = a1;
}

// ---------------------------------------------------------------------------
extern "C" void kernel_launch(void* const* d_in, const int* in_sizes, int n_in,
                              void* d_out, int out_size, void* d_ws, size_t ws_size,
                              hipStream_t stream) {
    const float* x    = (const float*)d_in[0];
    const int*   ei   = (const int*)d_in[1];
    const float* cW   = (const float*)d_in[2];
    const float* cb   = (const float*)d_in[3];
    const float* lncg = (const float*)d_in[4];
    const float* lncb = (const float*)d_in[5];
    const float* gW[2]  = {(const float*)d_in[6],  (const float*)d_in[12]};
    const float* gas[2] = {(const float*)d_in[7],  (const float*)d_in[13]};
    const float* gad[2] = {(const float*)d_in[8],  (const float*)d_in[14]};
    const float* gb[2]  = {(const float*)d_in[9],  (const float*)d_in[15]};
    const float* lng[2] = {(const float*)d_in[10], (const float*)d_in[16]};
    const float* lnb[2] = {(const float*)d_in[11], (const float*)d_in[17]};
    const float* mW1 = (const float*)d_in[18];
    const float* mb1 = (const float*)d_in[19];
    const float* mW2 = (const float*)d_in[20];
    const float* mb2 = (const float*)d_in[21];
    float* out = (float*)d_out;

    const int N = in_sizes[0] / IN_CH;
    const int E = in_sizes[1] / 2;

    char* p = (char*)d_ws;
    auto take = [&](size_t b) -> char* { char* r = p; p += (b + 255) & ~(size_t)255; return r; };
    __bf16* Wc   = (__bf16*)take((size_t)KPAD_C * HID * 2);
    __bf16* Wg0  = (__bf16*)take((size_t)HID * HID * 2);
    __bf16* Wg1  = (__bf16*)take((size_t)HID * HID * 2);
    __bf16* Wm1  = (__bf16*)take((size_t)HID * 64 * 2);
    float*  h    = (float*)take((size_t)N * HID * 4);
    __bf16* hbf  = (__bf16*)take((size_t)N * HID * 2);
    float*  hp   = (float*)take((size_t)N * HID * 4);
    float*  agg  = (float*)take((size_t)N * HID * 4);
    float*  als  = (float*)take((size_t)N * 8 * 4);
    float*  ald  = (float*)take((size_t)N * 8 * 4);
    float*  emax = (float*)take((size_t)N * 8 * 4);
    float*  denom= (float*)take((size_t)N * 8 * 4);
    float*  ebuf = (float*)take((size_t)E * 8 * 4);
    float*  z    = hp;  // hp dead after last GAT layer; reuse for MLP hidden

    // Weight repacks (bf16 fragment order)
    repack_kernel<<<((HID / 16) * (KPAD_C / 32) * 32 + 255) / 256, 256, 0, stream>>>(cW, Wc, QDIM, KPAD_C, HID);
    repack_kernel<<<((HID / 16) * (HID / 32) * 32 + 255) / 256, 256, 0, stream>>>(gW[0], Wg0, HID, HID, HID);
    repack_kernel<<<((HID / 16) * (HID / 32) * 32 + 255) / 256, 256, 0, stream>>>(gW[1], Wg1, HID, HID, HID);
    repack_kernel<<<((64 / 16) * (HID / 32) * 32 + 255) / 256, 256, 0, stream>>>(mW1, Wm1, HID, HID, 64);

    const int gblocks = (N + 15) / 16;

    // compress: h = elu(LN(quantum_map(x) @ Wc + b))   (qmap fused into A stage)
    wmma_gemm_kernel<true><<<gblocks, 256, 0, stream>>>(x, nullptr, Wc, cb, lncg, lncb,
                                                        h, hbf, N, KPAD_C, HID, 7);

    const __bf16* Wg[2] = {Wg0, Wg1};
    const int eblocks = (E * 8 + 255) / 256;
    for (int L = 0; L < 2; L++) {
        // hp = h @ W   (no bias/LN; bias applied after aggregation)
        wmma_gemm_kernel<false><<<gblocks, 256, 0, stream>>>(nullptr, hbf, Wg[L],
                                                             nullptr, nullptr, nullptr,
                                                             hp, nullptr, N, HID, HID, 0);
        att_scores_kernel<<<(N * 8 + 255) / 256, 256, 0, stream>>>(hp, gas[L], gad[L], als, ald, N);
        gat_init_kernel<<<(N * (HID + 16) + 255) / 256, 256, 0, stream>>>(agg, denom, emax, N);
        edge_logits_max_kernel<<<eblocks, 256, 0, stream>>>(ei, als, ald, emax, ebuf, E);
        edge_exp_sum_kernel<<<eblocks, 256, 0, stream>>>(ei, emax, denom, ebuf, E);
        edge_aggregate_kernel<<<eblocks, 256, 0, stream>>>(ei, hp, ebuf, denom, agg, E);
        gat_finalize_kernel<<<gblocks, 256, 0, stream>>>(agg, gb[L], lng[L], lnb[L], h, hbf, N);
    }

    // MLP: z = elu(h @ W1 + b1) ; out = z @ W2 + b2
    wmma_gemm_kernel<false><<<gblocks, 128, 0, stream>>>(nullptr, hbf, Wm1, mb1,
                                                         nullptr, nullptr, z, nullptr,
                                                         N, HID, 64, 5);
    mlp2_kernel<<<(N + 255) / 256, 256, 0, stream>>>(z, mW2, mb2, out, N);
}
